// GraphAttentionLayer_81810537054471
// MI455X (gfx1250) — matrix operand order repaired and checked by reading, not compiled, not run on hardware
//
#include <hip/hip_runtime.h>
#include <hip/hip_bf16.h>

#define NN 8192
#define FEAT 256
#define HID 64
#define ALPHA 0.2f

typedef __attribute__((ext_vector_type(16))) _Float16 v16h;
typedef __attribute__((ext_vector_type(8)))  float    v8f;

// ---------------- k0: zero accumulators ----------------
__global__ void gat_init(float* __restrict__ acc, float* __restrict__ denom,
                         unsigned* __restrict__ maxkey) {
  int i = blockIdx.x * blockDim.x + threadIdx.x;
  if (i < NN * HID) acc[i] = 0.0f;
  if (i < NN)       denom[i] = 0.0f;
  if (i == 0)       *maxkey = 0u;   // key of -inf
}

// ---------------- k1: h = X@W, s_src, s_dst, hT(f16), max(s_dst) ----------------
// one wave (32 lanes) per row; lane t owns hidden dims t and t+32
__global__ __launch_bounds__(32) void gat_h(const float* __restrict__ X,
                                            const float* __restrict__ W,
                                            const float* __restrict__ a,
                                            _Float16* __restrict__ hT,
                                            float* __restrict__ s_src,
                                            float* __restrict__ s_dst,
                                            unsigned* __restrict__ maxkey) {
  const int r = blockIdx.x;
  const int t = threadIdx.x;
  const float* xr = X + (size_t)r * FEAT;
  float acc0 = 0.0f, acc1 = 0.0f;
#pragma unroll 8
  for (int k = 0; k < FEAT; ++k) {
    float x = xr[k];
    acc0 = fmaf(x, W[k * HID + t],      acc0);
    acc1 = fmaf(x, W[k * HID + t + 32], acc1);
  }
  hT[(size_t)t * NN + r]        = (_Float16)acc0;
  hT[(size_t)(t + 32) * NN + r] = (_Float16)acc1;

  float ps = acc0 * a[t]      + acc1 * a[t + 32];
  float pd = acc0 * a[64 + t] + acc1 * a[96 + t];
#pragma unroll
  for (int off = 16; off > 0; off >>= 1) {
    ps += __shfl_down(ps, off);
    pd += __shfl_down(pd, off);
  }
  if (t == 0) {
    s_src[r] = ps;
    s_dst[r] = pd;
    unsigned u = __float_as_uint(pd);
    u = (u & 0x80000000u) ? ~u : (u | 0x80000000u);   // order-preserving key
    atomicMax(maxkey, u);
  }
}

// ---------------- k2: fused masked softmax + P@h via f16 WMMA ----------------
// grid = (KCHUNKS=16, ROWBLKS=256), 1 wave per block, 32 rows x 512 cols per block
__device__ __forceinline__ void make_p8(int4 ma, int4 mb, float4 sa, float4 sb,
                                        float ss, float mi, float& dsum,
                                        _Float16* o) {
  float s[8] = {sa.x, sa.y, sa.z, sa.w, sb.x, sb.y, sb.z, sb.w};
  int   m[8] = {ma.x, ma.y, ma.z, ma.w, mb.x, mb.y, mb.z, mb.w};
#pragma unroll
  for (int i = 0; i < 8; ++i) {
    float tv = ss + s[i];
    float lr = fmaxf(tv, ALPHA * tv);        // leaky-relu (alpha < 1)
    float p  = __expf(lr - mi);              // mi is a row upper bound -> p <= ~1
    p = (m[i] != 0) ? p : 0.0f;              // adjacency mask
    dsum += p;
    o[i] = (_Float16)p;
  }
}

__global__ __launch_bounds__(32) void gat_attn(const int* __restrict__ adj,
                                               const _Float16* __restrict__ hT,
                                               const float* __restrict__ s_src,
                                               const float* __restrict__ s_dst,
                                               const unsigned* __restrict__ maxkey,
                                               float* __restrict__ acc,
                                               float* __restrict__ denom) {
  __shared__ __align__(16) float sdst_sh[512];
  const int j0   = blockIdx.x * 512;     // K-chunk
  const int m0   = blockIdx.y * 32;      // row block (two 16-row WMMA blocks)
  const int lane = threadIdx.x;
  const int mrow = lane & 15;            // A: row M; B/C: column N
  const int half = lane >> 4;
  const int kb   = half * 8;             // A-operand K sub-group base

  for (int i = lane; i < 512; i += 32) sdst_sh[i] = s_dst[j0 + i];
  __syncthreads();

  unsigned u = *maxkey;
  float mdst = (u & 0x80000000u) ? __uint_as_float(u & 0x7fffffffu)
                                 : __uint_as_float(~u);
  const int r0 = m0 + mrow, r1 = m0 + 16 + mrow;
  const float ss0 = s_src[r0], ss1 = s_src[r1];
  float t0 = ss0 + mdst, t1 = ss1 + mdst;
  const float mi0 = fmaxf(t0, ALPHA * t0);   // row-wise softmax shift (upper bound)
  const float mi1 = fmaxf(t1, ALPHA * t1);

  const int* base0 = adj + (size_t)r0 * NN + j0;
  const int* base1 = adj + (size_t)r1 * NN + j0;

  v8f c0[4] = {v8f{}, v8f{}, v8f{}, v8f{}};
  v8f c1[4] = {v8f{}, v8f{}, v8f{}, v8f{}};
  float dsum0 = 0.0f, dsum1 = 0.0f;

  for (int step = 0; step < 16; ++step) {
    const int jk = step * 32;
    // adjacency: 2 rows x 2 K-groups x 8 ints (32B contiguous each)
    const int4* a0g0 = (const int4*)(base0 + jk + kb);
    const int4* a0g1 = (const int4*)(base0 + jk + kb + 16);
    const int4* a1g0 = (const int4*)(base1 + jk + kb);
    const int4* a1g1 = (const int4*)(base1 + jk + kb + 16);
    int4 m00a = a0g0[0], m00b = a0g0[1];
    int4 m01a = a0g1[0], m01b = a0g1[1];
    int4 m10a = a1g0[0], m10b = a1g0[1];
    int4 m11a = a1g1[0], m11b = a1g1[1];
    // s_dst from LDS (shared between both rows)
    const float4* s0 = (const float4*)&sdst_sh[jk + kb];
    const float4* s1 = (const float4*)&sdst_sh[jk + kb + 16];
    float4 sg0a = s0[0], sg0b = s0[1];
    float4 sg1a = s1[0], sg1b = s1[1];

    _Float16 pa[16], pb[16];
    make_p8(m00a, m00b, sg0a, sg0b, ss0, mi0, dsum0, pa);      // K = kb..kb+7
    make_p8(m01a, m01b, sg1a, sg1b, ss0, mi0, dsum0, pa + 8);  // K = kb+16..kb+23
    make_p8(m10a, m10b, sg0a, sg0b, ss1, mi1, dsum1, pb);
    make_p8(m11a, m11b, sg1a, sg1b, ss1, mi1, dsum1, pb + 8);
    v16h A0, A1;
#pragma unroll
    for (int i = 0; i < 16; ++i) { A0[i] = pa[i]; A1[i] = pb[i]; }

#pragma unroll
    for (int t = 0; t < 4; ++t) {
      // B tile: lane = column d = 16t+mrow; halves j = 8*half..+7 and 16+8*half..+7
      const _Float16* bq = hT + (size_t)(t * 16 + mrow) * NN + j0 + jk + 8 * half;
      union { uint4 q[2]; v16h v; } bu;
      bu.q[0] = *(const uint4*)bq;
      bu.q[1] = *(const uint4*)(bq + 16);
      v16h B = bu.v;
      c0[t] = __builtin_amdgcn_wmma_f32_16x16x32_f16(false, A0, false, B,
                                                     (short)0, c0[t], false, false);
      c1[t] = __builtin_amdgcn_wmma_f32_16x16x32_f16(false, A1, false, B,
                                                     (short)0, c1[t], false, false);
    }
  }

  // combine partial denominators (lane m holds K{kb,kb+16} groups; m+16 the rest)
  dsum0 += __shfl_xor(dsum0, 16);
  dsum1 += __shfl_xor(dsum1, 16);
  if (lane < 16) {
    atomicAdd(&denom[m0 + mrow],      dsum0);
    atomicAdd(&denom[m0 + 16 + mrow], dsum1);
  }
  // C layout: VGPR r -> row M = r + 8*half, column N = lane&15
#pragma unroll
  for (int t = 0; t < 4; ++t) {
#pragma unroll
    for (int r = 0; r < 8; ++r) {
      int col = t * 16 + mrow;
      atomicAdd(&acc[(size_t)(m0 + 8 * half + r) * HID + col],      c0[t][r]);
      atomicAdd(&acc[(size_t)(m0 + 16 + 8 * half + r) * HID + col], c1[t][r]);
    }
  }
}

// ---------------- k3: out = relu(acc / denom) ----------------
__global__ void gat_final(const float* __restrict__ acc,
                          const float* __restrict__ denom,
                          float* __restrict__ out) {
  int i = blockIdx.x * blockDim.x + threadIdx.x;
  if (i < NN * HID) {
    float d = denom[i / HID];
    float v = (d > 0.0f) ? (acc[i] / d) : 0.0f;
    out[i] = fmaxf(v, 0.0f);
  }
}

extern "C" void kernel_launch(void* const* d_in, const int* in_sizes, int n_in,
                              void* d_out, int out_size, void* d_ws, size_t ws_size,
                              hipStream_t stream) {
  const float* X   = (const float*)d_in[0];   // [8192,256]
  const int*   adj = (const int*)d_in[1];     // [8192,8192]
  // d_in[2] = tree_attention (unused by reference)
  const float* W   = (const float*)d_in[3];   // [256,64]
  const float* a   = (const float*)d_in[4];   // [128,1]
  float* out = (float*)d_out;

  char* w = (char*)d_ws;
  _Float16* hT     = (_Float16*)(w);                    // 1,048,576 B
  float*    acc    = (float*)(w + 1048576);             // 2,097,152 B
  float*    s_src  = (float*)(w + 1048576 + 2097152);   // 32 KB
  float*    s_dst  = (float*)(w + 1048576 + 2097152 + 32768);
  float*    denom  = (float*)(w + 1048576 + 2097152 + 65536);
  unsigned* maxkey = (unsigned*)(w + 1048576 + 2097152 + 98304);

  gat_init<<<(NN * HID + 255) / 256, 256, 0, stream>>>(acc, denom, maxkey);
  gat_h<<<NN, 32, 0, stream>>>(X, W, a, hT, s_src, s_dst, maxkey);
  gat_attn<<<dim3(16, 256), 32, 0, stream>>>(adj, hT, s_src, s_dst, maxkey, acc, denom);
  gat_final<<<(NN * HID + 255) / 256, 256, 0, stream>>>(acc, denom, out);
}
